// GlobalLinearSelfAttention_17575006175733
// MI455X (gfx1250) — matrix operand order repaired and checked by reading, not compiled, not run on hardware
//
#include <hip/hip_runtime.h>

// ---------------------------------------------------------------------------
// Problem constants (from reference)
// ---------------------------------------------------------------------------
#define BATCH   4
#define SEQ     4096
#define DMODEL  1024
#define HEADS   16
#define DHEAD   64
#define BN      (BATCH * SEQ)          // 16384 rows
#define QKVCOLS (3 * DMODEL)           // 3072

// ---------------------------------------------------------------------------
// WMMA vector types (CDNA5 / gfx1250, wave32)
// ---------------------------------------------------------------------------
typedef __attribute__((ext_vector_type(16))) __bf16 v16bf;
typedef __attribute__((ext_vector_type(8)))  float  v8f;
typedef __attribute__((ext_vector_type(4)))  int    v4i;

union FragBf { v16bf v; uint4 u4[2]; };

// CDNA5 async global->LDS copy path (ASYNCcnt-tracked), guarded so a missing
// builtin falls back to the manual global->VGPR->ds_store path.
#if __has_builtin(__builtin_amdgcn_global_load_async_to_lds_b128) && \
    __has_builtin(__builtin_amdgcn_s_wait_asynccnt)
#define HAVE_ASYNC_LDS 1
#else
#define HAVE_ASYNC_LDS 0
#endif

// ---------------------------------------------------------------------------
// f32 -> bf16 conversion (4 elems / thread)
// ---------------------------------------------------------------------------
__global__ __launch_bounds__(256)
void cvt_bf16_kernel(const float* __restrict__ src, __bf16* __restrict__ dst, long count) {
    long i = ((long)blockIdx.x * blockDim.x + threadIdx.x) * 4;
    if (i + 4 <= count) {
        float4 f = *(const float4*)(src + i);
        union { __bf16 b[4]; uint2 u; } o;
        o.b[0] = (__bf16)f.x; o.b[1] = (__bf16)f.y;
        o.b[2] = (__bf16)f.z; o.b[3] = (__bf16)f.w;
        *(uint2*)(dst + i) = o.u;
    }
}

// ---------------------------------------------------------------------------
// Generic batched GEMM:  C[M,N] (f32 or bf16) = A_bf16[M,K] @ B_bf16[K,N]
// Row-major, per-batch strides. Block = 256 threads = 8 waves (wave32).
// Block tile 128x128, K-step 32. Wave grid 4(M) x 2(N); each wave owns a
// 32x64 sub-tile = 2x4 accumulators of v_wmma_f32_16x16x32_bf16.
// A tile copied global->LDS with GLOBAL_LOAD_ASYNC_TO_LDS_B128 (ASYNCcnt);
// B tile is stored TRANSPOSED in LDS so B-fragments are contiguous reads.
// ---------------------------------------------------------------------------
__global__ __launch_bounds__(256)
void gemm_bf16_kernel(const __bf16* __restrict__ A, const __bf16* __restrict__ B,
                      void* __restrict__ C,
                      int M, int N, int K,
                      long ldA, long ldB, long ldC,
                      long sA, long sB, long sC,
                      int c_is_bf16) {
    __shared__ __bf16 As[128 * 32];   // [row][k]   8 KB
    __shared__ __bf16 Bs[128 * 32];   // [n][k]     8 KB (transposed)

    const int bz = blockIdx.z;
    A += (long)bz * sA;
    B += (long)bz * sB;

    const int n0 = blockIdx.x * 128;
    const int m0 = blockIdx.y * 128;

    const int tid   = threadIdx.x;
    const int lane  = tid & 31;
    const int wave  = tid >> 5;
    const int waveM = wave >> 1;      // 0..3
    const int waveN = wave & 1;       // 0..1
    const int lrow  = lane & 15;

    v8f acc[2][4];
#pragma unroll
    for (int mi = 0; mi < 2; ++mi)
#pragma unroll
        for (int ni = 0; ni < 4; ++ni)
            acc[mi][ni] = (v8f){0.f, 0.f, 0.f, 0.f, 0.f, 0.f, 0.f, 0.f};

    // A-fragment K pattern (ISA 7.12.2, 16-bit A 16x32):
    //   lanes 0-15 : K = 0..7, 16..23 ; lanes 16-31 : K = 8..15, 24..31
    const int kbA = (lane < 16) ? 0 : 8;
    // B-fragment (K x 16): lanes 0-15 hold K=0..7,16..23 ; lanes 16-31 hold K=8..15,24..31
    const int kbB = (lane < 16) ? 0 : 8;

    // per-thread tile-copy coordinates
    const int aRow = tid >> 1;                 // 0..127
    const int aCol = (tid & 1) << 4;           // 0 or 16
    const int bRow = tid >> 3;                 // 0..31
    const int bCol = (tid & 7) << 4;           // 0..112

    for (int k0 = 0; k0 < K; k0 += 32) {
        // ---- A tile: 128 rows x 32 halves, async DMA into LDS
        {
            __bf16* dst = As + aRow * 32 + aCol;
            if (m0 + aRow < M) {
                const __bf16* src = A + (long)(m0 + aRow) * ldA + k0 + aCol;
#if HAVE_ASYNC_LDS
                __builtin_amdgcn_global_load_async_to_lds_b128(
                    (v4i*)src, (v4i*)dst, 0, 0);
                __builtin_amdgcn_global_load_async_to_lds_b128(
                    (v4i*)(src + 8), (v4i*)(dst + 8), 0, 0);
#else
                *(uint4*)(dst)     = *(const uint4*)(src);
                *(uint4*)(dst + 8) = *(const uint4*)(src + 8);
#endif
                if (k0 + 32 < K)   // warm next K tile (global_prefetch_b8)
                    __builtin_prefetch(src + 32, 0, 3);
            } else {
                uint4 z = {0, 0, 0, 0};
                *(uint4*)(dst)     = z;
                *(uint4*)(dst + 8) = z;
            }
        }
        // ---- B tile (32 x 128): global->VGPR, stored transposed [n][k]
        {
            __bf16 tmp[16];
            if (n0 + bCol + 16 <= N) {
                const __bf16* src = B + (long)(k0 + bRow) * ldB + n0 + bCol;
                *(uint4*)(tmp)     = *(const uint4*)(src);
                *(uint4*)(tmp + 8) = *(const uint4*)(src + 8);
                if (k0 + 32 < K)
                    __builtin_prefetch(src + 32 * ldB, 0, 3);
            } else {
#pragma unroll
                for (int i = 0; i < 16; ++i) tmp[i] = (__bf16)0.f;
            }
#pragma unroll
            for (int i = 0; i < 16; ++i)
                Bs[(bCol + i) * 32 + bRow] = tmp[i];
        }
#if HAVE_ASYNC_LDS
        __builtin_amdgcn_s_wait_asynccnt(0);   // my async LDS writes landed
#endif
        __syncthreads();

        // ---- fragments + WMMA
        FragBf a[2], b[4];
#pragma unroll
        for (int mi = 0; mi < 2; ++mi) {
            const __bf16* p = As + (waveM * 32 + mi * 16 + lrow) * 32;
            a[mi].u4[0] = *(const uint4*)(p + kbA);
            a[mi].u4[1] = *(const uint4*)(p + kbA + 16);
        }
#pragma unroll
        for (int ni = 0; ni < 4; ++ni) {
            const __bf16* p = Bs + (waveN * 64 + ni * 16 + lrow) * 32 + kbB;
            b[ni].u4[0] = *(const uint4*)(p);
            b[ni].u4[1] = *(const uint4*)(p + 16);
        }
#pragma unroll
        for (int mi = 0; mi < 2; ++mi)
#pragma unroll
            for (int ni = 0; ni < 4; ++ni)
                acc[mi][ni] = __builtin_amdgcn_wmma_f32_16x16x32_bf16(
                    false, a[mi].v, false, b[ni].v,
                    (short)0, acc[mi][ni], false, false);
        __syncthreads();
    }

    // ---- epilogue: C/D layout (lane<16: M=r, N=lane ; lane>=16: M=8+r)
    const int rAdd = (lane < 16) ? 0 : 8;
#pragma unroll
    for (int mi = 0; mi < 2; ++mi) {
#pragma unroll
        for (int ni = 0; ni < 4; ++ni) {
            const int gc = n0 + waveN * 64 + ni * 16 + lrow;
#pragma unroll
            for (int r = 0; r < 8; ++r) {
                const int gr = m0 + waveM * 32 + mi * 16 + rAdd + r;
                if (gr < M && gc < N) {
                    float val = acc[mi][ni][r];
                    if (c_is_bf16)
                        ((__bf16*)C)[(long)bz * sC + (long)gr * ldC + gc] = (__bf16)val;
                    else
                        ((float*)C)[(long)bz * sC + (long)gr * ldC + gc] = val;
                }
            }
        }
    }
}

// ---------------------------------------------------------------------------
// Row softmax over d (64) for Q (scaled by d^-1/2), plus V bf16 re-layout.
// One wave per (b, n, h) row; outputs in per-head layout [(b*H+h)*SEQ + n][d].
// ---------------------------------------------------------------------------
__global__ __launch_bounds__(256)
void softmax_q_and_v_kernel(const float* __restrict__ qkv,
                            __bf16* __restrict__ qBf, __bf16* __restrict__ vBf) {
    const int lane = threadIdx.x & 31;
    const int wave = threadIdx.x >> 5;
    const long row = (long)blockIdx.x * 8 + wave;   // (b*SEQ+n)*HEADS + h
    const long bn  = row >> 4;                      // b*SEQ + n
    const int  h   = (int)(row & 15);
    const long b   = bn >> 12;                      // SEQ = 4096
    const long n   = bn & 4095;

    const float* src = qkv + bn * QKVCOLS + h * DHEAD;
    float x0 = src[lane], x1 = src[lane + 32];

    float m = fmaxf(x0, x1);
#pragma unroll
    for (int off = 16; off > 0; off >>= 1)
        m = fmaxf(m, __shfl_xor(m, off, 32));
    float e0 = __expf(x0 - m), e1 = __expf(x1 - m);
    float s = e0 + e1;
#pragma unroll
    for (int off = 16; off > 0; off >>= 1)
        s += __shfl_xor(s, off, 32);
    const float inv = 0.125f / s;                   // scale = DHEAD^-0.5

    const long dstRow = ((b * HEADS + h) * (long)SEQ + n) * DHEAD;
    qBf[dstRow + lane]      = (__bf16)(e0 * inv);
    qBf[dstRow + lane + 32] = (__bf16)(e1 * inv);

    const float* vs = qkv + bn * QKVCOLS + 2 * DMODEL + h * DHEAD;
    vBf[dstRow + lane]      = (__bf16)vs[lane];
    vBf[dstRow + lane + 32] = (__bf16)vs[lane + 32];
}

// ---------------------------------------------------------------------------
// Column softmax over n (4096) for K; one workgroup per (b,h).
// Output written TRANSPOSED as kT[(b*H+h)*64 + d][n] (bf16) so the context
// GEMM's A-matrix is row-major. Transposed stores staged through LDS so the
// global writes are contiguous 32 B chunks.
// ---------------------------------------------------------------------------
__global__ __launch_bounds__(256)
void softmax_k_kernel(const float* __restrict__ qkv, __bf16* __restrict__ kT) {
    const int bh = blockIdx.x;
    const int b  = bh >> 4;
    const int h  = bh & 15;
    const int tid = threadIdx.x;
    const int d = tid & 63;
    const int g = tid >> 6;                         // 0..3 threads per column

    const float* base = qkv + (long)b * SEQ * QKVCOLS + DMODEL + h * DHEAD + d;

    __shared__ float red[4][64];
    __shared__ __bf16 tbuf[64 * 72];                // padded: 16B-aligned chunks

    // pass 1: column max
    float m = -3.4e38f;
    for (int n = g; n < SEQ; n += 4)
        m = fmaxf(m, base[(long)n * QKVCOLS]);
    red[g][d] = m;
    __syncthreads();
    m = fmaxf(fmaxf(red[0][d], red[1][d]), fmaxf(red[2][d], red[3][d]));
    __syncthreads();

    // pass 2: column sum of exp
    float s = 0.f;
    for (int n = g; n < SEQ; n += 4)
        s += __expf(base[(long)n * QKVCOLS] - m);
    red[g][d] = s;
    __syncthreads();
    s = red[0][d] + red[1][d] + red[2][d] + red[3][d];
    const float inv = 1.f / s;
    __syncthreads();

    // pass 3: normalize + transpose via LDS, 64-n chunks
    __bf16* dst = kT + (long)bh * DHEAD * SEQ;
    for (int nb = 0; nb < SEQ; nb += 64) {
#pragma unroll
        for (int j = 0; j < 16; ++j) {
            const int nl = g * 16 + j;
            float v = __expf(base[(long)(nb + nl) * QKVCOLS] - m) * inv;
            tbuf[d * 72 + nl] = (__bf16)v;
        }
        __syncthreads();
        const int row = tid >> 2;                   // d index 0..63
        const int cb  = (tid & 3) << 4;             // 0,16,32,48
        uint4 w0 = *(const uint4*)(tbuf + row * 72 + cb);
        uint4 w1 = *(const uint4*)(tbuf + row * 72 + cb + 8);
        *(uint4*)(dst + (long)row * SEQ + nb + cb)     = w0;
        *(uint4*)(dst + (long)row * SEQ + nb + cb + 8) = w1;
        __syncthreads();
    }
}

// ---------------------------------------------------------------------------
// Merge heads: [(b*H+h)][n][e] -> [(b*SEQ+n)][h*64+e]  (bf16, 8 elems/thread)
// ---------------------------------------------------------------------------
__global__ __launch_bounds__(256)
void merge_heads_kernel(const __bf16* __restrict__ src, __bf16* __restrict__ dst) {
    const long i  = ((long)blockIdx.x * blockDim.x + threadIdx.x) * 8; // dst idx
    const int  e  = (int)(i & 63);
    const int  h  = (int)((i >> 6) & 15);
    const long bn = i >> 10;
    const long b  = bn >> 12;
    const long n  = bn & 4095;
    const long srcIdx = ((b * HEADS + h) * (long)SEQ + n) * DHEAD + e;
    *(uint4*)(dst + i) = *(const uint4*)(src + srcIdx);
}

// ---------------------------------------------------------------------------
// Host-side orchestration
// ---------------------------------------------------------------------------
extern "C" void kernel_launch(void* const* d_in, const int* in_sizes, int n_in,
                              void* d_out, int out_size, void* d_ws, size_t ws_size,
                              hipStream_t stream) {
    const float* feats = (const float*)d_in[0];   // [4,4096,1024]
    const float* w_qkv = (const float*)d_in[1];   // [1024,3072]
    const float* w_out = (const float*)d_in[2];   // [1024,1024]
    float* out = (float*)d_out;                   // [4,4096,1024]

    char* ws = (char*)d_ws;
    size_t off = 0;
    auto alloc = [&](size_t bytes) -> char* {
        char* p = ws + off;
        off += (bytes + 255) & ~(size_t)255;
        return p;
    };

    __bf16* featsBf    = (__bf16*)alloc((size_t)BN * DMODEL * 2);
    __bf16* wqkvBf     = (__bf16*)alloc((size_t)DMODEL * QKVCOLS * 2);
    __bf16* woutBf     = (__bf16*)alloc((size_t)DMODEL * DMODEL * 2);
    float*  qkvF       = (float*) alloc((size_t)BN * QKVCOLS * 4);
    __bf16* qBf        = (__bf16*)alloc((size_t)BN * DMODEL * 2);  // per-head layout
    __bf16* kTBf       = (__bf16*)alloc((size_t)BN * DMODEL * 2);  // per-head, transposed [d][n]
    __bf16* vBf        = (__bf16*)alloc((size_t)BN * DMODEL * 2);  // per-head layout
    __bf16* ctxBf      = (__bf16*)alloc((size_t)BATCH * HEADS * DHEAD * DHEAD * 2);
    __bf16* attnHeadBf = (__bf16*)alloc((size_t)BN * DMODEL * 2);  // per-head layout
    __bf16* attnBf     = (__bf16*)alloc((size_t)BN * DMODEL * 2);  // merged [bn][h*64+e]

    // 1) f32 -> bf16 conversions
    cvt_bf16_kernel<<<(BN * DMODEL) / 1024, 256, 0, stream>>>(feats, featsBf, (long)BN * DMODEL);
    cvt_bf16_kernel<<<(DMODEL * QKVCOLS) / 1024, 256, 0, stream>>>(w_qkv, wqkvBf, (long)DMODEL * QKVCOLS);
    cvt_bf16_kernel<<<(DMODEL * DMODEL) / 1024, 256, 0, stream>>>(w_out, woutBf, (long)DMODEL * DMODEL);

    // 2) qkv = feats @ w_qkv   (M=16384, N=3072, K=1024), f32 out for softmax
    gemm_bf16_kernel<<<dim3(QKVCOLS / 128, BN / 128, 1), 256, 0, stream>>>(
        featsBf, wqkvBf, qkvF, BN, QKVCOLS, DMODEL,
        DMODEL, QKVCOLS, QKVCOLS, 0, 0, 0, /*c_bf16=*/0);

    // 3) softmax(q over d) * scale, re-layout v  (one wave per row)
    softmax_q_and_v_kernel<<<(BN * HEADS) / 8, 256, 0, stream>>>(qkvF, qBf, vBf);

    // 4) softmax(k over n), written transposed per head
    softmax_k_kernel<<<BATCH * HEADS, 256, 0, stream>>>(qkvF, kTBf);

    // 5) context[bh] = kT[64,4096] @ v[4096,64]  (batch=64), bf16 out
    gemm_bf16_kernel<<<dim3(1, 1, BATCH * HEADS), 256, 0, stream>>>(
        kTBf, vBf, ctxBf, DHEAD, DHEAD, SEQ,
        SEQ, DHEAD, DHEAD,
        (long)DHEAD * SEQ, (long)SEQ * DHEAD, (long)DHEAD * DHEAD, /*c_bf16=*/1);

    // 6) out_head[bh] = q[4096,64] @ context[64,64]  (batch=64), bf16 out
    gemm_bf16_kernel<<<dim3(1, SEQ / 128, BATCH * HEADS), 256, 0, stream>>>(
        qBf, ctxBf, attnHeadBf, SEQ, DHEAD, DHEAD,
        DHEAD, DHEAD, DHEAD,
        (long)SEQ * DHEAD, (long)DHEAD * DHEAD, (long)SEQ * DHEAD, /*c_bf16=*/1);

    // 7) merge heads back to [bn][1024]
    merge_heads_kernel<<<(BN * DMODEL) / (256 * 8), 256, 0, stream>>>(attnHeadBf, attnBf);

    // 8) out = attn @ w_out  (M=16384, N=1024, K=1024), f32 to d_out
    gemm_bf16_kernel<<<dim3(DMODEL / 128, BN / 128, 1), 256, 0, stream>>>(
        attnBf, woutBf, out, BN, DMODEL, DMODEL,
        DMODEL, DMODEL, DMODEL, 0, 0, 0, /*c_bf16=*/0);
}
